// EncodeProcessDecode_84164179133070
// MI455X (gfx1250) — matrix operand order repaired and checked by reading, not compiled, not run on hardware
//
#include <hip/hip_runtime.h>

// CDNA5 / gfx1250 wave32 WMMA implementation of EncodeProcessDecode GraphNet.
// All matmuls (encoder/core/decoder MLPs) run on v_wmma_f32_16x16x32_f16:
// 16 rows (edges/nodes) x 16 outputs x K<=32 per instruction, f32 accumulate.

typedef __attribute__((ext_vector_type(16))) _Float16 v16h;
typedef __attribute__((ext_vector_type(8)))  float    v8f;

#define N_NODES 100000
#define N_EDGES 1600000
#define T_STEPS 5

// Wave-local LDS ordering: ds ops complete in order; wait them out.
// Wave executes in lockstep so no barrier is needed for cross-lane LDS.
__device__ __forceinline__ void wave_lds_sync() {
  asm volatile("s_wait_dscnt 0x0" ::: "memory");
}

__device__ __forceinline__ v8f wmma16(v16h a, v16h b, v8f c) {
  // (neg_a, A, neg_b, B, c_mod, C, reuse_a, reuse_b)
  return __builtin_amdgcn_wmma_f32_16x16x32_f16(false, a, false, b, (short)0, c,
                                                false, false);
}

// B fragment (K x N, K<=32, N<=16), row-major W[k*N + n].
// Lane n (0..15) holds column n for K=0..7,16..23; lane n+16 holds K=8..15,24..31.
__device__ __forceinline__ v16h pack_B(const float* __restrict__ W, int K, int N,
                                       int col, bool hi) {
  v16h r;
#pragma unroll
  for (int i = 0; i < 16; ++i) {
    int k = (i < 8 ? i : i + 8) + (hi ? 8 : 0);
    float v = (k < K && col < N) ? W[k * N + col] : 0.f;
    r[i] = (_Float16)v;
  }
  return r;
}

// B with every column equal to w[0..15] (K=16): D[m][n] = dot(A_row_m, w) for all n.
__device__ __forceinline__ v16h pack_B_rep(const float* __restrict__ w, bool hi) {
  v16h r;
#pragma unroll
  for (int i = 0; i < 16; ++i) r[i] = (_Float16)(i < 8 ? w[i + (hi ? 8 : 0)] : 0.f);
  return r;
}

// A fragment row from a 16-float LDS row (K<=16; K=16..31 zero).
// A layout (16-bit, 16x32): lanes 0-15 hold K 0-7 / 16-23, lanes 16-31 hold K 8-15 / 24-31.
__device__ __forceinline__ v16h pack_A_lds(const float* row, bool hi) {
  v16h r;
  if (!hi) {
#pragma unroll
    for (int i = 0; i < 8; ++i) { r[i] = (_Float16)row[i];     r[i + 8] = (_Float16)0.f; }
  } else {
#pragma unroll
    for (int i = 0; i < 8; ++i) { r[i] = (_Float16)row[8 + i]; r[i + 8] = (_Float16)0.f; }
  }
  return r;
}

__global__ void zero_kernel(float* p, int n) {
  int i = blockIdx.x * blockDim.x + threadIdx.x;
  if (i < n) p[i] = 0.f;
}

__global__ void count_kernel(const int* __restrict__ edges, float* __restrict__ cnt) {
  int i = blockIdx.x * blockDim.x + threadIdx.x;
  if (i < N_EDGES) atomicAdd(&cnt[edges[N_EDGES + i]], 1.f);
}

// ---------------- encoder: edge 4->16->5 -> e0, e ----------------
__global__ void __launch_bounds__(256)
enc_edge_kernel(const float* __restrict__ attr,
                const float* __restrict__ W1, const float* __restrict__ b1,
                const float* __restrict__ W2, const float* __restrict__ b2,
                float* __restrict__ e0, float* __restrict__ e) {
  __shared__ float lds[8][256];
  const int lane = threadIdx.x & 31, wave = threadIdx.x >> 5;
  const int col = lane & 15;
  const bool hi = lane >= 16;
  float* h = lds[wave];
  v16h Bw1 = pack_B(W1, 4, 16, col, hi);
  v16h Bw2 = pack_B(W2, 16, 5, col, hi);
  float bias1 = b1[col];
  float bias2 = (col < 5) ? b2[col] : 0.f;
  const int waveBase = (blockIdx.x * 8 + wave) * 32;
#pragma unroll
  for (int tile = 0; tile < 2; ++tile) {
    const int base = waveBase + tile * 16;
    const int idx = base + col;
    v16h A;
#pragma unroll
    for (int i = 0; i < 16; ++i) A[i] = (_Float16)0.f;
    if (!hi) {  // K=0..3
      A[0] = (_Float16)attr[idx * 4 + 0]; A[1] = (_Float16)attr[idx * 4 + 1];
      A[2] = (_Float16)attr[idx * 4 + 2]; A[3] = (_Float16)attr[idx * 4 + 3];
    }
    v8f z = {};
    v8f D1 = wmma16(A, Bw1, z);
    wave_lds_sync();
#pragma unroll
    for (int j = 0; j < 8; ++j) h[(j + (hi ? 8 : 0)) * 16 + col] = fmaxf(D1[j] + bias1, 0.f);
    wave_lds_sync();
    v16h A2 = pack_A_lds(&h[col * 16], hi);
    v8f D2 = wmma16(A2, Bw2, z);
    if (col < 5) {
#pragma unroll
      for (int j = 0; j < 8; ++j) {
        int m = base + j + (hi ? 8 : 0);
        float v = D2[j] + bias2;
        e0[m * 5 + col] = v;
        e[m * 5 + col] = v;
      }
    }
  }
}

// ---------------- encoder: node 8->16->5 -> vcomb=[v0,v0] ----------------
__global__ void __launch_bounds__(256)
enc_node_kernel(const float* __restrict__ attr,
                const float* __restrict__ W1, const float* __restrict__ b1,
                const float* __restrict__ W2, const float* __restrict__ b2,
                float* __restrict__ vcomb) {
  __shared__ float lds[8][256];
  const int lane = threadIdx.x & 31, wave = threadIdx.x >> 5;
  const int wgid = blockIdx.x * 8 + wave;
  if (wgid >= N_NODES / 32) return;  // 3125 waves
  const int col = lane & 15;
  const bool hi = lane >= 16;
  float* h = lds[wave];
  v16h Bw1 = pack_B(W1, 8, 16, col, hi);
  v16h Bw2 = pack_B(W2, 16, 5, col, hi);
  float bias1 = b1[col];
  float bias2 = (col < 5) ? b2[col] : 0.f;
  const int waveBase = wgid * 32;
#pragma unroll
  for (int tile = 0; tile < 2; ++tile) {
    const int base = waveBase + tile * 16;
    const int idx = base + col;
    v16h A;
#pragma unroll
    for (int i = 0; i < 16; ++i) A[i] = (_Float16)0.f;
    if (!hi) {  // K=0..7
#pragma unroll
      for (int i = 0; i < 8; ++i) A[i] = (_Float16)attr[idx * 8 + i];
    }
    v8f z = {};
    v8f D1 = wmma16(A, Bw1, z);
    wave_lds_sync();
#pragma unroll
    for (int j = 0; j < 8; ++j) h[(j + (hi ? 8 : 0)) * 16 + col] = fmaxf(D1[j] + bias1, 0.f);
    wave_lds_sync();
    v16h A2 = pack_A_lds(&h[col * 16], hi);
    v8f D2 = wmma16(A2, Bw2, z);
    if (col < 5) {
#pragma unroll
      for (int j = 0; j < 8; ++j) {
        int m = base + j + (hi ? 8 : 0);
        float v = D2[j] + bias2;
        vcomb[m * 10 + col] = v;      // v0
        vcomb[m * 10 + 5 + col] = v;  // v (== v0 at t=0)
      }
    }
  }
}

// ------------- core edge: [e0,e,vc[src],vc[dst]](30)->16->5, scatter, dec 5->16->1 -------------
__global__ void __launch_bounds__(256)
edge_core_kernel(const float* __restrict__ e0, float* __restrict__ e,
                 const float* __restrict__ vcomb, const int* __restrict__ edges,
                 float* __restrict__ agg,
                 const float* __restrict__ W1, const float* __restrict__ b1,
                 const float* __restrict__ W2, const float* __restrict__ b2,
                 const float* __restrict__ dW1, const float* __restrict__ db1,
                 const float* __restrict__ dW2, const float* __restrict__ db2,
                 float* __restrict__ out_e) {
  __shared__ float lds[8][256];
  const int lane = threadIdx.x & 31, wave = threadIdx.x >> 5;
  const int col = lane & 15;
  const bool hi = lane >= 16;
  float* h = lds[wave];
  v16h Bw1 = pack_B(W1, 30, 16, col, hi);
  v16h Bw2 = pack_B(W2, 16, 5, col, hi);
  v16h Bd1 = pack_B(dW1, 5, 16, col, hi);
  v16h Bd2 = pack_B_rep(dW2, hi);  // dec_e_W2: 16x1
  float bias1 = b1[col];
  float bias2 = (col < 5) ? b2[col] : 0.f;
  float biasd1 = db1[col];
  float biasd2 = db2[0];
  const int waveBase = (blockIdx.x * 8 + wave) * 32;
#pragma unroll
  for (int tile = 0; tile < 2; ++tile) {
    const int base = waveBase + tile * 16;
    const int idx = base + col;
    const int src = edges[idx];
    const int dst = edges[N_EDGES + idx];
    const float* vs = vcomb + src * 10;
    const float* vd = vcomb + dst * 10;
    const float* pe0 = e0 + idx * 5;
    const float* pe = e + idx * 5;
    // A row = [e0(5), e(5), vc[src](10), vc[dst](10), 0, 0]; each lane loads its half of K.
    v16h A;
    if (!hi) {  // K 0-7, 16-23
      A[0] = (_Float16)pe0[0]; A[1] = (_Float16)pe0[1]; A[2] = (_Float16)pe0[2];
      A[3] = (_Float16)pe0[3]; A[4] = (_Float16)pe0[4];
      A[5] = (_Float16)pe[0];  A[6] = (_Float16)pe[1];  A[7] = (_Float16)pe[2];
      A[8]  = (_Float16)vs[6]; A[9]  = (_Float16)vs[7];
      A[10] = (_Float16)vs[8]; A[11] = (_Float16)vs[9];
      A[12] = (_Float16)vd[0]; A[13] = (_Float16)vd[1];
      A[14] = (_Float16)vd[2]; A[15] = (_Float16)vd[3];
    } else {    // K 8-15, 24-31
      A[0] = (_Float16)pe[3];  A[1] = (_Float16)pe[4];
      A[2] = (_Float16)vs[0];  A[3] = (_Float16)vs[1];  A[4] = (_Float16)vs[2];
      A[5] = (_Float16)vs[3];  A[6] = (_Float16)vs[4];  A[7] = (_Float16)vs[5];
      A[8]  = (_Float16)vd[4]; A[9]  = (_Float16)vd[5]; A[10] = (_Float16)vd[6];
      A[11] = (_Float16)vd[7]; A[12] = (_Float16)vd[8]; A[13] = (_Float16)vd[9];
      A[14] = (_Float16)0.f;   A[15] = (_Float16)0.f;
    }
    v8f z = {};
    v8f D1 = wmma16(A, Bw1, z);  // hidden 16
    wave_lds_sync();
#pragma unroll
    for (int j = 0; j < 8; ++j) h[(j + (hi ? 8 : 0)) * 16 + col] = fmaxf(D1[j] + bias1, 0.f);
    wave_lds_sync();
    v16h A2 = pack_A_lds(&h[col * 16], hi);
    v8f D2 = wmma16(A2, Bw2, z);  // e_new (cols 0-4; cols>=5 exact 0)
    wave_lds_sync();
#pragma unroll
    for (int j = 0; j < 8; ++j) h[(j + (hi ? 8 : 0)) * 16 + col] = D2[j] + bias2;
    if (col < 5) {
#pragma unroll
      for (int j = 0; j < 8; ++j) {
        int m = base + j + (hi ? 8 : 0);
        float v = D2[j] + bias2;
        e[m * 5 + col] = v;
        atomicAdd(&agg[edges[N_EDGES + m] * 5 + col], v);
      }
    }
    wave_lds_sync();
    v16h A3 = pack_A_lds(&h[col * 16], hi);
    v8f D3 = wmma16(A3, Bd1, z);  // decoder hidden
    wave_lds_sync();
#pragma unroll
    for (int j = 0; j < 8; ++j) h[(j + (hi ? 8 : 0)) * 16 + col] = fmaxf(D3[j] + biasd1, 0.f);
    wave_lds_sync();
    v16h A4 = pack_A_lds(&h[col * 16], hi);
    v8f D4 = wmma16(A4, Bd2, z);  // every column = scalar output
    if (col == 0) {
#pragma unroll
      for (int j = 0; j < 8; ++j) out_e[base + j + (hi ? 8 : 0)] = D4[j] + biasd2;
    }
  }
}

// ------------- core node: [vc(10), mean_agg(5)]->16->5, dec 5->16->1 -------------
__global__ void __launch_bounds__(256)
node_core_kernel(float* __restrict__ vcomb, const float* __restrict__ agg,
                 const float* __restrict__ cnt,
                 const float* __restrict__ W1, const float* __restrict__ b1,
                 const float* __restrict__ W2, const float* __restrict__ b2,
                 const float* __restrict__ dW1, const float* __restrict__ db1,
                 const float* __restrict__ dW2, const float* __restrict__ db2,
                 float* __restrict__ out_v) {
  __shared__ float lds[8][256];
  const int lane = threadIdx.x & 31, wave = threadIdx.x >> 5;
  const int wgid = blockIdx.x * 8 + wave;
  if (wgid >= N_NODES / 32) return;
  const int col = lane & 15;
  const bool hi = lane >= 16;
  float* h = lds[wave];
  v16h Bw1 = pack_B(W1, 15, 16, col, hi);
  v16h Bw2 = pack_B(W2, 16, 5, col, hi);
  v16h Bd1 = pack_B(dW1, 5, 16, col, hi);
  v16h Bd2 = pack_B_rep(dW2, hi);
  float bias1 = b1[col];
  float bias2 = (col < 5) ? b2[col] : 0.f;
  float biasd1 = db1[col];
  float biasd2 = db2[0];
  const int waveBase = wgid * 32;
#pragma unroll
  for (int tile = 0; tile < 2; ++tile) {
    const int base = waveBase + tile * 16;
    const int node = base + col;
    const float* vc = vcomb + node * 10;
    v16h A;
#pragma unroll
    for (int i = 0; i < 16; ++i) A[i] = (_Float16)0.f;
    if (!hi) {  // K 0-7 = vc[0..7]
#pragma unroll
      for (int i = 0; i < 8; ++i) A[i] = (_Float16)vc[i];
    } else {    // K 8,9 = vc[8,9]; K 10-14 = mean agg
      float c = fmaxf(cnt[node], 1.f);
      A[0] = (_Float16)vc[8];
      A[1] = (_Float16)vc[9];
#pragma unroll
      for (int i = 0; i < 5; ++i) A[2 + i] = (_Float16)(agg[node * 5 + i] / c);
    }
    v8f z = {};
    v8f D1 = wmma16(A, Bw1, z);
    wave_lds_sync();
#pragma unroll
    for (int j = 0; j < 8; ++j) h[(j + (hi ? 8 : 0)) * 16 + col] = fmaxf(D1[j] + bias1, 0.f);
    wave_lds_sync();
    v16h A2 = pack_A_lds(&h[col * 16], hi);
    v8f D2 = wmma16(A2, Bw2, z);  // v_new
    wave_lds_sync();
#pragma unroll
    for (int j = 0; j < 8; ++j) h[(j + (hi ? 8 : 0)) * 16 + col] = D2[j] + bias2;
    if (col < 5) {
#pragma unroll
      for (int j = 0; j < 8; ++j) {
        int m = base + j + (hi ? 8 : 0);
        vcomb[m * 10 + 5 + col] = D2[j] + bias2;  // update v half of vcomb
      }
    }
    wave_lds_sync();
    v16h A3 = pack_A_lds(&h[col * 16], hi);
    v8f D3 = wmma16(A3, Bd1, z);
    wave_lds_sync();
#pragma unroll
    for (int j = 0; j < 8; ++j) h[(j + (hi ? 8 : 0)) * 16 + col] = fmaxf(D3[j] + biasd1, 0.f);
    wave_lds_sync();
    v16h A4 = pack_A_lds(&h[col * 16], hi);
    v8f D4 = wmma16(A4, Bd2, z);
    if (col == 0) {
#pragma unroll
      for (int j = 0; j < 8; ++j) out_v[base + j + (hi ? 8 : 0)] = D4[j] + biasd2;
    }
  }
}

// ------------- global feature chain (tiny) -------------
__global__ void u_kernel(const float* __restrict__ u, const float* __restrict__ Wu,
                         const float* __restrict__ bu, const float* __restrict__ dWu,
                         const float* __restrict__ dbu, float* __restrict__ out_u) {
  if (threadIdx.x == 0 && blockIdx.x == 0) {
    float u0 = u[0];
    float uu = u0;
    for (int t = 0; t < T_STEPS; ++t) {
      uu = u0 * Wu[0] + uu * Wu[1] + bu[0];
      out_u[t] = uu * dWu[0] + dbu[0];
    }
  }
}

extern "C" void kernel_launch(void* const* d_in, const int* in_sizes, int n_in,
                              void* d_out, int out_size, void* d_ws, size_t ws_size,
                              hipStream_t stream) {
  const float* edge_attr = (const float*)d_in[0];
  const float* node_attr = (const float*)d_in[1];
  const float* u         = (const float*)d_in[2];
  const int*   edges     = (const int*)d_in[3];
  // d_in[4] = num_steps (compile-time T_STEPS per reference)
  const float* enc_e_W1 = (const float*)d_in[5],  *enc_e_b1 = (const float*)d_in[6];
  const float* enc_e_W2 = (const float*)d_in[7],  *enc_e_b2 = (const float*)d_in[8];
  const float* enc_v_W1 = (const float*)d_in[9],  *enc_v_b1 = (const float*)d_in[10];
  const float* enc_v_W2 = (const float*)d_in[11], *enc_v_b2 = (const float*)d_in[12];
  const float* core_e_W1 = (const float*)d_in[13], *core_e_b1 = (const float*)d_in[14];
  const float* core_e_W2 = (const float*)d_in[15], *core_e_b2 = (const float*)d_in[16];
  const float* core_v_W1 = (const float*)d_in[17], *core_v_b1 = (const float*)d_in[18];
  const float* core_v_W2 = (const float*)d_in[19], *core_v_b2 = (const float*)d_in[20];
  const float* core_u_W = (const float*)d_in[21], *core_u_b = (const float*)d_in[22];
  const float* dec_e_W1 = (const float*)d_in[23], *dec_e_b1 = (const float*)d_in[24];
  const float* dec_e_W2 = (const float*)d_in[25], *dec_e_b2 = (const float*)d_in[26];
  const float* dec_v_W1 = (const float*)d_in[27], *dec_v_b1 = (const float*)d_in[28];
  const float* dec_v_W2 = (const float*)d_in[29], *dec_v_b2 = (const float*)d_in[30];
  const float* dec_u_W = (const float*)d_in[31], *dec_u_b = (const float*)d_in[32];

  // workspace: e0[E*5], e[E*5], vcomb[N*10]=[v0|v], agg[N*5], cnt[N]  (~70.4 MB)
  float* e0    = (float*)d_ws;
  float* e     = e0 + (size_t)N_EDGES * 5;
  float* vcomb = e + (size_t)N_EDGES * 5;
  float* agg   = vcomb + (size_t)N_NODES * 10;
  float* cnt   = agg + (size_t)N_NODES * 5;

  float* out_e = (float*)d_out;                            // [T, E, 1]
  float* out_v = out_e + (size_t)T_STEPS * N_EDGES;        // [T, N, 1]
  float* out_u = out_v + (size_t)T_STEPS * N_NODES;        // [T, 1, 1]

  const int nodeBlocks = (N_NODES / 32 + 7) / 8;  // 391

  zero_kernel<<<(N_NODES + 255) / 256, 256, 0, stream>>>(cnt, N_NODES);
  count_kernel<<<(N_EDGES + 255) / 256, 256, 0, stream>>>(edges, cnt);
  enc_edge_kernel<<<N_EDGES / 256, 256, 0, stream>>>(edge_attr, enc_e_W1, enc_e_b1,
                                                     enc_e_W2, enc_e_b2, e0, e);
  enc_node_kernel<<<nodeBlocks, 256, 0, stream>>>(node_attr, enc_v_W1, enc_v_b1,
                                                  enc_v_W2, enc_v_b2, vcomb);
  for (int t = 0; t < T_STEPS; ++t) {
    zero_kernel<<<(N_NODES * 5 + 255) / 256, 256, 0, stream>>>(agg, N_NODES * 5);
    edge_core_kernel<<<N_EDGES / 256, 256, 0, stream>>>(
        e0, e, vcomb, edges, agg,
        core_e_W1, core_e_b1, core_e_W2, core_e_b2,
        dec_e_W1, dec_e_b1, dec_e_W2, dec_e_b2,
        out_e + (size_t)t * N_EDGES);
    node_core_kernel<<<nodeBlocks, 256, 0, stream>>>(
        vcomb, agg, cnt,
        core_v_W1, core_v_b1, core_v_W2, core_v_b2,
        dec_v_W1, dec_v_b1, dec_v_W2, dec_v_b2,
        out_v + (size_t)t * N_NODES);
  }
  u_kernel<<<1, 1, 0, stream>>>(u, core_u_W, core_u_b, dec_u_W, dec_u_b, out_u);
}
